// FlexAttentionWithRoPE_81913616269887
// MI455X (gfx1250) — matrix operand order repaired and checked by reading, not compiled, not run on hardware
//
#include <hip/hip_runtime.h>

#define BB 2
#define LL 2048
#define HH 16
#define HD 64
#define CC 1024
#define C3 3072
#define ML (BB * LL) // 4096 rows

typedef __attribute__((ext_vector_type(16))) __bf16 v16bf;
typedef __attribute__((ext_vector_type(8)))  float  v8f;
typedef __attribute__((ext_vector_type(4)))  int    v4i;

union ABFrag { v16bf bf; v4i part[2]; };

__device__ __forceinline__ unsigned short f2bf(float f) {
  unsigned u = __float_as_uint(f);
  u += 0x7FFFu + ((u >> 16) & 1u);   // round-to-nearest-even
  return (unsigned short)(u >> 16);
}

__device__ __forceinline__ v8f wmma_bf16(const ABFrag& a, const ABFrag& b, v8f c) {
  // (neg_a, A, neg_b, B, c_mod, C, reuse_a, reuse_b)
  return __builtin_amdgcn_wmma_f32_16x16x32_bf16(false, a.bf, false, b.bf,
                                                 (short)0, c, false, false);
}

__device__ __forceinline__ void load_a_frag(ABFrag& f, const unsigned short* row, int half) {
  f.part[0] = *(const v4i*)(row + 8 * half);
  f.part[1] = *(const v4i*)(row + 16 + 8 * half);
}
__device__ __forceinline__ void load_b_frag(ABFrag& f, const unsigned short* row, int half) {
  f.part[0] = *(const v4i*)(row + 8 * half);
  f.part[1] = *(const v4i*)(row + 16 + 8 * half);
}

// ---------------- conversion / transpose ----------------
__global__ __launch_bounds__(256)
void cast_f32_to_bf16(const float* __restrict__ src, unsigned short* __restrict__ dst, int n) {
  int i = blockIdx.x * blockDim.x + threadIdx.x;
  if (i < n) dst[i] = f2bf(src[i]);
}

// src: rows x cols row-major fp32; dst: cols x rows bf16 (K-major for WMMA B operand)
__global__ __launch_bounds__(256)
void transpose_to_bf16(const float* __restrict__ src, unsigned short* __restrict__ dst,
                       int rows, int cols) {
  int i = blockIdx.x * blockDim.x + threadIdx.x;
  if (i < rows * cols) {
    int k = i % rows;   // dst inner (K)
    int n = i / rows;   // dst outer (N)
    dst[i] = f2bf(src[(size_t)k * cols + n]);
  }
}

// ---------------- bf16 WMMA GEMM: C[M,N] = A[M,K] * BT[N,K]^T + bias ----------------
// Register-blocked 32x64 per wave: 2 M-tiles x 4 N-tiles, 8 WMMAs per K=32 step,
// 12 b128 loads per step (A frags reused 4x, B frags reused 2x).
__global__ __launch_bounds__(128)
void gemm_bf16_rowcol(const unsigned short* __restrict__ A,
                      const unsigned short* __restrict__ BT,
                      const float* __restrict__ bias,
                      float* __restrict__ Cout, int K, int N) {
  int wave = threadIdx.x >> 5, lane = threadIdx.x & 31;
  int half = lane >> 4, lidx = lane & 15;
  int ntiles = N >> 6;                       // 64-wide N tiles
  int tile = blockIdx.x * 4 + wave;
  int mt = tile / ntiles, nt = tile - mt * ntiles;
  int m0 = mt << 5, n0 = nt << 6;

  const unsigned short* arow0 = A + (size_t)(m0 + lidx) * K;
  const unsigned short* arow1 = A + (size_t)(m0 + 16 + lidx) * K;
  const unsigned short* brow0 = BT + (size_t)(n0 + lidx) * K;
  const unsigned short* brow1 = BT + (size_t)(n0 + 16 + lidx) * K;
  const unsigned short* brow2 = BT + (size_t)(n0 + 32 + lidx) * K;
  const unsigned short* brow3 = BT + (size_t)(n0 + 48 + lidx) * K;

  v8f acc[2][4];
#pragma unroll
  for (int mi = 0; mi < 2; ++mi)
#pragma unroll
    for (int ni = 0; ni < 4; ++ni) acc[mi][ni] = (v8f){};

  for (int k0 = 0; k0 < K; k0 += 32) {
    ABFrag a0, a1, b0, b1, b2, b3;
    load_a_frag(a0, arow0 + k0, half);
    load_a_frag(a1, arow1 + k0, half);
    load_b_frag(b0, brow0 + k0, half);
    load_b_frag(b1, brow1 + k0, half);
    load_b_frag(b2, brow2 + k0, half);
    load_b_frag(b3, brow3 + k0, half);
    acc[0][0] = wmma_bf16(a0, b0, acc[0][0]);
    acc[0][1] = wmma_bf16(a0, b1, acc[0][1]);
    acc[0][2] = wmma_bf16(a0, b2, acc[0][2]);
    acc[0][3] = wmma_bf16(a0, b3, acc[0][3]);
    acc[1][0] = wmma_bf16(a1, b0, acc[1][0]);
    acc[1][1] = wmma_bf16(a1, b1, acc[1][1]);
    acc[1][2] = wmma_bf16(a1, b2, acc[1][2]);
    acc[1][3] = wmma_bf16(a1, b3, acc[1][3]);
  }

#pragma unroll
  for (int mi = 0; mi < 2; ++mi) {
#pragma unroll
    for (int ni = 0; ni < 4; ++ni) {
      float bi = bias[n0 + 16 * ni + lidx];
      float* out = Cout + (size_t)(m0 + 16 * mi + 8 * half) * N + (n0 + 16 * ni + lidx);
#pragma unroll
      for (int r = 0; r < 8; ++r) out[(size_t)r * N] = acc[mi][ni][r] + bi;
    }
  }
}

// ---------------- RoPE + repack: qkv fp32 -> q_rot/k_rot [bh][l][d], vT [bh][d][l] ----------------
__global__ __launch_bounds__(256)
void rope_pack(const float* __restrict__ qkv, const float* __restrict__ freqs,
               unsigned short* __restrict__ qr, unsigned short* __restrict__ kr,
               unsigned short* __restrict__ vT) {
  int wave = threadIdx.x >> 5, lane = threadIdx.x & 31;
  int row = blockIdx.x * 8 + wave;           // (b*H + h)*L + l
  int l = row & (LL - 1);
  int bh = row >> 11;
  int h = bh & (HH - 1), b = bh >> 4;
  const float* base = qkv + ((size_t)(b * LL + l)) * C3 + h * HD;
  float q1 = base[lane],          q2 = base[lane + 32];
  float k1 = base[CC + lane],     k2 = base[CC + lane + 32];
  float v1 = base[2 * CC + lane], v2 = base[2 * CC + lane + 32];
  float th = (float)l * freqs[h * 32 + lane];
  float s, c;
  __sincosf(th, &s, &c);
  unsigned short* qrow = qr + ((size_t)bh * LL + l) * HD;
  unsigned short* krow = kr + ((size_t)bh * LL + l) * HD;
  qrow[lane]      = f2bf(q1 * c - q2 * s);
  qrow[lane + 32] = f2bf(q1 * s + q2 * c);
  krow[lane]      = f2bf(k1 * c - k2 * s);
  krow[lane + 32] = f2bf(k1 * s + k2 * c);
  unsigned short* vb = vT + (size_t)bh * HD * LL;
  vb[(size_t)lane * LL + l]        = f2bf(v1);
  vb[(size_t)(lane + 32) * LL + l] = f2bf(v2);
}

// ---------------- flash attention: one wave per 32-query tile ----------------
// Per 32-key chunk: 16 WMMAs fed by 16 b128 fragment loads (K/V frags reused by
// both query sub-tiles) -> 1 load per WMMA, and K/V L2 traffic halved vs 16-query tiles.
__global__ __launch_bounds__(128)
void attention(const unsigned short* __restrict__ qr,
               const unsigned short* __restrict__ kr,
               const unsigned short* __restrict__ vT,
               unsigned short* __restrict__ ao) {
  __shared__ __align__(16) unsigned short pbuf[4][2][16 * 32];   // 8 KB
  int wave = threadIdx.x >> 5, lane = threadIdx.x & 31;
  int half = lane >> 4, lidx = lane & 15;
  int tile = blockIdx.x * 4 + wave;
  int qt = tile & 63;               // L/32 = 64 query tiles (32 queries each)
  int bh = tile >> 6;
  int b = bh >> 4, h = bh & 15;

  const unsigned short* Qbase = qr + ((size_t)bh * LL + qt * 32) * HD;
  const unsigned short* Kbase = kr + (size_t)bh * LL * HD;
  const unsigned short* Vbase = vT + (size_t)bh * HD * LL;

  // Q A-fragments: 2 query sub-tiles x 2 dim-chunks (reused for all key chunks)
  ABFrag Aq[2][2];
#pragma unroll
  for (int qi = 0; qi < 2; ++qi) {
    const unsigned short* qrow = Qbase + (size_t)(16 * qi + lidx) * HD;
    load_a_frag(Aq[qi][0], qrow, half);
    load_a_frag(Aq[qi][1], qrow + 32, half);
  }

  float mrow[2][8], lrow[2][8];
#pragma unroll
  for (int qi = 0; qi < 2; ++qi)
#pragma unroll
    for (int r = 0; r < 8; ++r) { mrow[qi][r] = -1e30f; lrow[qi][r] = 0.f; }
  v8f O[2][4];
#pragma unroll
  for (int qi = 0; qi < 2; ++qi)
#pragma unroll
    for (int t = 0; t < 4; ++t) O[qi][t] = (v8f){};

  for (int kc = 0; kc < LL; kc += 32) {
    // prefetch next chunk (global_prefetch_b8; no LOADcnt cost)
    if (kc + 32 < LL) {
      __builtin_prefetch(Kbase + (size_t)(kc + 32 + lidx) * HD, 0, 3);
      __builtin_prefetch(Kbase + (size_t)(kc + 48 + lidx) * HD, 0, 3);
      __builtin_prefetch(Vbase + (size_t)lidx * LL + kc + 32, 0, 3);
    }

    // K B-fragments: lane = key column; contiguous dim-runs (shared by both q sub-tiles)
    const unsigned short* krow0 = Kbase + (size_t)(kc + lidx) * HD;
    const unsigned short* krow1 = Kbase + (size_t)(kc + 16 + lidx) * HD;
    ABFrag B00, B01, B10, B11;
    load_b_frag(B00, krow0, half);
    load_b_frag(B01, krow0 + 32, half);
    load_b_frag(B10, krow1, half);
    load_b_frag(B11, krow1 + 32, half);

    float alpha[2][8];
#pragma unroll
    for (int qi = 0; qi < 2; ++qi) {
      v8f S0 = {}, S1 = {};
      S0 = wmma_bf16(Aq[qi][0], B00, S0);
      S0 = wmma_bf16(Aq[qi][1], B01, S0);
      S1 = wmma_bf16(Aq[qi][0], B10, S1);
      S1 = wmma_bf16(Aq[qi][1], B11, S1);

      // online softmax over this 32-key chunk; lane holds rows 8*half+r, one key col
      float nm[8];
#pragma unroll
      for (int r = 0; r < 8; ++r) {
        S0[r] *= 0.125f; S1[r] *= 0.125f;    // 1/sqrt(64)
        nm[r] = fmaxf(S0[r], S1[r]);
      }
#pragma unroll
      for (int mask = 1; mask < 16; mask <<= 1)
#pragma unroll
        for (int r = 0; r < 8; ++r)
          nm[r] = fmaxf(nm[r], __shfl_xor(nm[r], mask, 32));

      float rs[8];
#pragma unroll
      for (int r = 0; r < 8; ++r) {
        float mn = fmaxf(mrow[qi][r], nm[r]);
        alpha[qi][r] = __expf(mrow[qi][r] - mn);
        mrow[qi][r] = mn;
        S0[r] = __expf(S0[r] - mn);
        S1[r] = __expf(S1[r] - mn);
        rs[r] = S0[r] + S1[r];
      }
#pragma unroll
      for (int mask = 1; mask < 16; mask <<= 1)
#pragma unroll
        for (int r = 0; r < 8; ++r)
          rs[r] += __shfl_xor(rs[r], mask, 32);
#pragma unroll
      for (int r = 0; r < 8; ++r)
        lrow[qi][r] = lrow[qi][r] * alpha[qi][r] + rs[r];

      // stage P (16x32 bf16) through LDS (wave-private slab): C-layout -> A-layout
      unsigned short* myP = pbuf[wave][qi];
#pragma unroll
      for (int r = 0; r < 8; ++r) {
        myP[(8 * half + r) * 32 + lidx]      = f2bf(S0[r]);
        myP[(8 * half + r) * 32 + 16 + lidx] = f2bf(S1[r]);
      }
    }

    __builtin_amdgcn_wave_barrier();
    asm volatile("s_wait_dscnt 0" ::: "memory");
    ABFrag Ap[2];
#pragma unroll
    for (int qi = 0; qi < 2; ++qi) {
      const unsigned short* prow = pbuf[wave][qi] + lidx * 32;
      Ap[qi].part[0] = *(const v4i*)(prow + 8 * half);
      Ap[qi].part[1] = *(const v4i*)(prow + 16 + 8 * half);
    }
    __builtin_amdgcn_wave_barrier();

#pragma unroll
    for (int qi = 0; qi < 2; ++qi)
#pragma unroll
      for (int t = 0; t < 4; ++t)
#pragma unroll
        for (int r = 0; r < 8; ++r)
          O[qi][t][r] *= alpha[qi][r];

    // PV: B = vT[dim][key]; each V fragment reused by both q sub-tiles
#pragma unroll
    for (int t = 0; t < 4; ++t) {
      ABFrag Bv;
      load_b_frag(Bv, Vbase + (size_t)(t * 16 + lidx) * LL + kc, half);
      O[0][t] = wmma_bf16(Ap[0], Bv, O[0][t]);
      O[1][t] = wmma_bf16(Ap[1], Bv, O[1][t]);
    }
  }

  // normalize and write bf16 to ao[b*L + l][h*64 + d] (A operand for final projection)
#pragma unroll
  for (int qi = 0; qi < 2; ++qi) {
    int rowbase = b * LL + qt * 32 + 16 * qi + 8 * half;
    int colbase = h * HD + lidx;
#pragma unroll
    for (int r = 0; r < 8; ++r) {
      float inv = 1.f / lrow[qi][r];
      size_t o = (size_t)(rowbase + r) * CC + colbase;
      ao[o +  0] = f2bf(O[qi][0][r] * inv);
      ao[o + 16] = f2bf(O[qi][1][r] * inv);
      ao[o + 32] = f2bf(O[qi][2][r] * inv);
      ao[o + 48] = f2bf(O[qi][3][r] * inv);
    }
  }
}

// ---------------- host-side orchestration ----------------
extern "C" void kernel_launch(void* const* d_in, const int* in_sizes, int n_in,
                              void* d_out, int out_size, void* d_ws, size_t ws_size,
                              hipStream_t stream) {
  const float* x      = (const float*)d_in[0];
  const float* w_qkv  = (const float*)d_in[1];
  const float* b_qkv  = (const float*)d_in[2];
  const float* w_out  = (const float*)d_in[3];
  const float* b_out  = (const float*)d_in[4];
  const float* rope   = (const float*)d_in[5];
  float* out = (float*)d_out;

  char* ws = (char*)d_ws;
  float*          qkv_f32 = (float*)ws;          ws += (size_t)ML * C3 * 4;
  unsigned short* xb      = (unsigned short*)ws; ws += (size_t)ML * CC * 2;
  unsigned short* wqT     = (unsigned short*)ws; ws += (size_t)C3 * CC * 2;
  unsigned short* woT     = (unsigned short*)ws; ws += (size_t)CC * CC * 2;
  unsigned short* qr      = (unsigned short*)ws; ws += (size_t)BB * HH * LL * HD * 2;
  unsigned short* kr      = (unsigned short*)ws; ws += (size_t)BB * HH * LL * HD * 2;
  unsigned short* vT      = (unsigned short*)ws; ws += (size_t)BB * HH * HD * LL * 2;
  unsigned short* ao      = (unsigned short*)ws; ws += (size_t)ML * CC * 2;

  cast_f32_to_bf16<<<(ML * CC + 255) / 256, 256, 0, stream>>>(x, xb, ML * CC);
  transpose_to_bf16<<<(CC * C3 + 255) / 256, 256, 0, stream>>>(w_qkv, wqT, CC, C3);
  transpose_to_bf16<<<(CC * CC + 255) / 256, 256, 0, stream>>>(w_out, woT, CC, CC);

  // qkv = x @ w_qkv + b ; 128 (32-row) M-tiles x 48 (64-col) N-tiles, 4 waves/block
  gemm_bf16_rowcol<<<(ML / 32) * (C3 / 64) / 4, 128, 0, stream>>>(xb, wqT, b_qkv, qkv_f32, CC, C3);

  rope_pack<<<BB * HH * LL / 8, 256, 0, stream>>>(qkv_f32, rope, qr, kr, vT);

  // 32 bh * 64 q-tiles = 2048 waves, 4 per block
  attention<<<BB * HH * (LL / 32) / 4, 128, 0, stream>>>(qr, kr, vT, ao);

  // out = ao @ w_out + b ; 128 x 16 tiles
  gemm_bf16_rowcol<<<(ML / 32) * (CC / 64) / 4, 128, 0, stream>>>(ao, woT, b_out, out, CC, CC);
}
